// LIFBlock_17454747091639
// MI455X (gfx1250) — compile-verified, matrix operand order checked by reading, not run on hardware
//
#include <hip/hip_runtime.h>
#include <hip/hip_bf16.h>
#include <math.h>

// ---------------------------------------------------------------------------
// LIFBlock for MI455X (gfx1250):
//  * bf16 WMMA (f32 acc) for every 1x1 conv
//  * global_load_async_to_lds_b128 (ASYNCcnt) for HBM->LDS activation staging
//  * fragment-ordered LDS so WMMA operand loads are 2x ds_load_b128
//  * branch-free GELU (v_rcp/v_exp based erf)
//  * fused GN prologues/epilogues, fused MLP with hidden tensor in LDS
// ---------------------------------------------------------------------------

typedef __attribute__((ext_vector_type(16))) __bf16 v16bf;
typedef __attribute__((ext_vector_type(8)))  float  v8f;
typedef unsigned short ushort_t;

#define C_   96
#define HW_  3136          // 56*56
#define B_   64
#define NPB  (C_ * HW_)    // 301056 elements per batch sample
#define HD_  384           // 4*C
#define EPS_ 1e-5f

__device__ __forceinline__ ushort_t f2bf(float f) {
  union { float f; unsigned u; } v; v.f = f;
  unsigned r = v.u + 0x7FFFu + ((v.u >> 16) & 1u);   // round-to-nearest-even
  return (ushort_t)(r >> 16);
}

// Branch-free exact-ish GELU: 0.5*(x + |x|*erf(|x|/sqrt(2))),
// erf via Abramowitz-Stegun 7.1.26 (|err| < 1.5e-7), v_rcp_f32 + v_exp_f32.
__device__ __forceinline__ float gelu_f(float x) {
  const float ax = fabsf(x) * 0.70710678118654752f;
  const float t  = __builtin_amdgcn_rcpf(fmaf(0.3275911f, ax, 1.0f));
  float p = fmaf(1.061405429f, t, -1.453152027f);
  p = fmaf(p, t, 1.421413741f);
  p = fmaf(p, t, -0.284496736f);
  p = fmaf(p, t, 0.254829592f);
  p *= t;
  const float ex = __builtin_amdgcn_exp2f(-ax * ax * 1.4426950408889634f);
  const float erfv = 1.0f - p * ex;
  return 0.5f * (x + fabsf(x) * erfv);
}

__device__ __forceinline__ void load_stats(const float* __restrict__ s, int b,
                                           float& mean, float& rstd) {
  float sum = s[b * 2 + 0], ssq = s[b * 2 + 1];
  mean = sum / (float)NPB;
  float var = ssq / (float)NPB - mean * mean;
  rstd = rsqrtf(var + EPS_);
}

// ---- CDNA5 async HBM->LDS copy (ASYNCcnt path) ----------------------------
__device__ __forceinline__ void async_b128(unsigned lds_off, const void* g) {
  asm volatile("global_load_async_to_lds_b128 %0, %1, off"
               :: "v"(lds_off), "v"((unsigned long long)(uintptr_t)g)
               : "memory");
}
__device__ __forceinline__ void wait_async0() {
  asm volatile("s_wait_asynccnt 0x0" ::: "memory");
}
// Copy a [96 x 64] f32 tile (rows strided by HW_) into LDS. 1536 16B chunks,
// 12 per thread with 128 threads.
__device__ __forceinline__ void stage_tile_async(const float* __restrict__ src,
                                                 unsigned lds_base, int tid) {
  #pragma unroll
  for (int i = 0; i < 12; ++i) {
    const int ch  = tid + i * 128;
    const int c   = ch >> 4;               // 16 chunks per 64-float row
    const int col = (ch & 15) << 2;        // float column
    async_b128(lds_base + (unsigned)(ch << 4),
               src + (size_t)c * HW_ + col);
  }
}

// Fragment index helpers (ISA 7.12.2, 16-bit A/B 16x32):
//   element e of lane L covers k = 8*(L>>4) + (e<8 ? e : e+8)
// inverse: kloc in [0,32): half=(kloc>>3)&1, e=(kloc&7)+(kloc&16 ? 8:0)
__device__ __forceinline__ int frag_e(int kloc)    { return (kloc & 7) + ((kloc & 16) ? 8 : 0); }
__device__ __forceinline__ int frag_half(int kloc) { return (kloc >> 3) & 1; }

union Frag { v16bf v; ushort_t s[16]; uint4 q[2]; };

// Vectorized A-fragment straight from global f32 weights: two runs of 8
// contiguous k values -> 4x global_load_b128 + cvt.
__device__ __forceinline__ v16bf load_a_frag_global(const float* __restrict__ W,
                                                    int row, int ldk, int kb, int half) {
  const float* p = W + (size_t)row * ldk + kb + half * 8;
  const float4 f0 = *(const float4*)(p);
  const float4 f1 = *(const float4*)(p + 4);
  const float4 f2 = *(const float4*)(p + 16);
  const float4 f3 = *(const float4*)(p + 20);
  Frag u;
  u.s[0]=f2bf(f0.x); u.s[1]=f2bf(f0.y); u.s[2]=f2bf(f0.z); u.s[3]=f2bf(f0.w);
  u.s[4]=f2bf(f1.x); u.s[5]=f2bf(f1.y); u.s[6]=f2bf(f1.z); u.s[7]=f2bf(f1.w);
  u.s[8]=f2bf(f2.x); u.s[9]=f2bf(f2.y); u.s[10]=f2bf(f2.z); u.s[11]=f2bf(f2.w);
  u.s[12]=f2bf(f3.x); u.s[13]=f2bf(f3.y); u.s[14]=f2bf(f3.z); u.s[15]=f2bf(f3.w);
  return u.v;
}
__device__ __forceinline__ v16bf load_frag_lds(const ushort_t* __restrict__ base, int off) {
  Frag u;
  u.q[0] = *(const uint4*)(base + off);
  u.q[1] = *(const uint4*)(base + off + 8);
  return u.v;
}

// --------------------------- stats helpers ---------------------------------
__global__ void zero_stats_kernel(float* __restrict__ p, int n) {
  for (int i = threadIdx.x; i < n; i += blockDim.x) p[i] = 0.f;
}

// grid (1176, B), block 256 : 1176*256 == NPB exactly
__global__ void batch_stats_kernel(const float* __restrict__ in,
                                   float* __restrict__ stats) {
  const int b = blockIdx.y;
  const int e = blockIdx.x * 256 + threadIdx.x;
  const float v = in[(size_t)b * NPB + e];
  __shared__ float red[512];
  red[threadIdx.x] = v; red[256 + threadIdx.x] = v * v;
  __syncthreads();
  for (int st = 128; st > 0; st >>= 1) {
    if (threadIdx.x < st) {
      red[threadIdx.x]       += red[threadIdx.x + st];
      red[256 + threadIdx.x] += red[256 + threadIdx.x + st];
    }
    __syncthreads();
  }
  if (threadIdx.x == 0) {
    atomicAdd(&stats[b * 2 + 0], red[0]);
    atomicAdd(&stats[b * 2 + 1], red[256]);
  }
}

// --------------------------- WMMA 1x1 conv ---------------------------------
// Block: 128 threads (4 waves), 64-pixel tile, all 96 output channels.
// Activations: async HBM->LDS raw f32, then GN+cvt into fragment-ordered bf16.
// Single-input: weights staged to fragment-ordered LDS (overlaps async copy).
// DUAL: out = gelu(W1 in1 + b1) + gelu(W2 in2 + b2), weights from global.
template<bool NORM, bool DUAL, bool RESID, bool STATS>
__global__ void conv1x1_wmma_kernel(
    const float* __restrict__ in1, const float* __restrict__ w1, const float* __restrict__ b1,
    const float* __restrict__ in2, const float* __restrict__ w2, const float* __restrict__ b2,
    const float* __restrict__ statsIn, const float* __restrict__ gamma, const float* __restrict__ beta,
    const float* __restrict__ resid,
    float* __restrict__ out, float* __restrict__ statsOut)
{
  constexpr int SMEM_BYTES = DUAL ? (24576 + 24576 + 1024) : (24576 + 18432 + 12288 + 1024);
  __shared__ __align__(16) unsigned char smem[SMEM_BYTES];
  float*    sRawF = (float*)smem;                                   // 24 KB raw f32 tile
  ushort_t* sW    = (ushort_t*)(smem + 24576);                      // single: 18 KB wfrag
  ushort_t* sIn   = (ushort_t*)(smem + (DUAL ? 24576 : 43008));     // frag bf16 (12/24 KB)
  float*    sRed  = (float*)(smem + (DUAL ? 49152 : 55296));

  const int tid = threadIdx.x;
  const int b   = blockIdx.y;
  const int p0  = blockIdx.x * 64;
  const size_t base = (size_t)b * NPB;
  const unsigned ldsRaw = (unsigned)(uintptr_t)sRawF;

  float mean = 0.f, rstd = 1.f;
  if constexpr (NORM) load_stats(statsIn, b, mean, rstd);

  // Kick off async copy of input tile, overlap weight staging with it.
  stage_tile_async(in1 + base + p0, ldsRaw, tid);

  if constexpr (!DUAL) {
    // stage weights into fragment order: [mt(6)][kt(3)][lane(32)][e(16)]
    for (int idx = tid; idx < C_ * C_; idx += 128) {
      const int m = idx / C_, k = idx - m * C_;
      const int mt = m >> 4, mr = m & 15;
      const int kt = k >> 5, kloc = k & 31;
      const int lane = frag_half(kloc) * 16 + mr;
      sW[(((mt * 3 + kt) * 32 + lane) << 4) + frag_e(kloc)] = f2bf(w1[idx]);
    }
  }
  wait_async0();
  __syncthreads();

  // Convert raw f32 -> GN -> bf16 fragment order: [pt(4)][kt(3)][lane][e]
  for (int idx = tid; idx < C_ * 64; idx += 128) {
    const int c = idx >> 6, p = idx & 63;
    float v = sRawF[idx];
    if constexpr (NORM) v = (v - mean) * rstd * gamma[c] + beta[c];
    const int kt = c >> 5, kloc = c & 31;
    const int lane = frag_half(kloc) * 16 + (p & 15);
    sIn[((((p >> 4) * 3 + kt) * 32 + lane) << 4) + frag_e(kloc)] = f2bf(v);
  }
  __syncthreads();

  if constexpr (DUAL) {                       // second input, reuse raw region
    stage_tile_async(in2 + base + p0, ldsRaw, tid);
    wait_async0();
    __syncthreads();
    for (int idx = tid; idx < C_ * 64; idx += 128) {
      const int c = idx >> 6, p = idx & 63;
      const int kt = c >> 5, kloc = c & 31;
      const int lane = frag_half(kloc) * 16 + (p & 15);
      sIn[6144 + ((((p >> 4) * 3 + kt) * 32 + lane) << 4) + frag_e(kloc)] = f2bf(sRawF[idx]);
    }
    __syncthreads();
  }

  const int wave = tid >> 5, lane = tid & 31;
  const int n = lane & 15, half = lane >> 4;

  float ls = 0.f, lss = 0.f;
  for (int mt = 0; mt < 6; ++mt) {
    v8f acc1 = {}; v8f acc2 = {};
    for (int ks = 0; ks < 3; ++ks) {
      const int boff = ((wave * 3 + ks) * 32 + lane) << 4;
      const v16bf bf1 = load_frag_lds(sIn, boff);
      v16bf af1;
      if constexpr (DUAL) af1 = load_a_frag_global(w1, mt * 16 + n, C_, ks * 32, half);
      else                af1 = load_frag_lds(sW, ((mt * 3 + ks) * 32 + lane) << 4);
      acc1 = __builtin_amdgcn_wmma_f32_16x16x32_bf16(false, af1, false, bf1,
                                                     (short)0, acc1, false, false);
      if constexpr (DUAL) {
        const v16bf bf2 = load_frag_lds(sIn, 6144 + boff);
        const v16bf af2 = load_a_frag_global(w2, mt * 16 + n, C_, ks * 32, half);
        acc2 = __builtin_amdgcn_wmma_f32_16x16x32_bf16(false, af2, false, bf2,
                                                       (short)0, acc2, false, false);
      }
    }
    #pragma unroll
    for (int r = 0; r < 8; ++r) {
      const int oc = mt * 16 + half * 8 + r;                    // D row mapping
      const size_t g = base + (size_t)oc * HW_ + p0 + wave * 16 + n;
      float v;
      if constexpr (DUAL) v = gelu_f(acc1[r] + b1[oc]) + gelu_f(acc2[r] + b2[oc]);
      else                v = acc1[r] + b1[oc];
      if constexpr (RESID) v += resid[g];
      out[g] = v;
      if constexpr (STATS) { ls += v; lss += v * v; }
    }
  }

  if constexpr (STATS) {
    sRed[tid] = ls; __syncthreads();
    for (int st = 64; st > 0; st >>= 1) { if (tid < st) sRed[tid] += sRed[tid + st]; __syncthreads(); }
    const float bs = sRed[0];
    __syncthreads();
    sRed[tid] = lss; __syncthreads();
    for (int st = 64; st > 0; st >>= 1) { if (tid < st) sRed[tid] += sRed[tid + st]; __syncthreads(); }
    if (tid == 0) {
      atomicAdd(&statsOut[b * 2 + 0], bs);
      atomicAdd(&statsOut[b * 2 + 1], sRed[0]);
    }
  }
}

// --------------------------- depthwise 3x3 ---------------------------------
__global__ void dwconv3_kernel(const float* __restrict__ in, const float* __restrict__ wt,
                               const float* __restrict__ bias,
                               const float* __restrict__ statsIn,
                               const float* __restrict__ gamma, const float* __restrict__ beta,
                               float* __restrict__ out, float* __restrict__ statsOut)
{
  const int b = blockIdx.y;
  const int e = blockIdx.x * 256 + threadIdx.x;        // [0, NPB)
  const int c = e / HW_, p = e % HW_;
  const int h = p / 56, w = p % 56;
  float mean, rstd; load_stats(statsIn, b, mean, rstd);
  const float gm = gamma[c], bt = beta[c];
  const float* src = in + (size_t)b * NPB + (size_t)c * HW_;
  float acc = 0.f;
  #pragma unroll
  for (int dh = -1; dh <= 1; ++dh) {
    #pragma unroll
    for (int dw = -1; dw <= 1; ++dw) {
      const int hh = h + dh, ww = w + dw;
      if (hh >= 0 && hh < 56 && ww >= 0 && ww < 56) {
        float v = src[hh * 56 + ww];
        v = gelu_f((v - mean) * rstd * gm + bt);
        acc += v * wt[c * 9 + (dh + 1) * 3 + (dw + 1)];
      }
    }
  }
  acc += bias[c];
  out[(size_t)b * NPB + e] = acc;

  __shared__ float red[512];
  red[threadIdx.x] = acc; red[256 + threadIdx.x] = acc * acc;
  __syncthreads();
  for (int st = 128; st > 0; st >>= 1) {
    if (threadIdx.x < st) {
      red[threadIdx.x]       += red[threadIdx.x + st];
      red[256 + threadIdx.x] += red[256 + threadIdx.x + st];
    }
    __syncthreads();
  }
  if (threadIdx.x == 0) {
    atomicAdd(&statsOut[b * 2 + 0], red[0]);
    atomicAdd(&statsOut[b * 2 + 1], red[256]);
  }
}

// --------------------------- LIF recurrence --------------------------------
template<bool ALONG_H>
__global__ void lif_kernel(const float* __restrict__ in,
                           const float* __restrict__ statsIn,
                           const float* __restrict__ gamma, const float* __restrict__ beta,
                           const float* __restrict__ tauP, const float* __restrict__ vthP,
                           float* __restrict__ out)
{
  const int idx = blockIdx.x * 256 + threadIdx.x;
  if (idx >= B_ * C_ * 56) return;
  const int fix = idx % 56;
  const int c = (idx / 56) % C_;
  const int b = idx / (56 * C_);
  float mean, rstd; load_stats(statsIn, b, mean, rstd);
  const float gm = gamma[c], bt = beta[c];
  const float tau = tauP[0], vth = vthP[0];
  const size_t base = (size_t)b * NPB + (size_t)c * HW_;
  float u = 0.f, o = 0.f;
  for (int t = 0; t < 56; ++t) {
    const size_t g = ALONG_H ? base + (size_t)(t * 56 + fix)
                             : base + (size_t)(fix * 56 + t);
    const float xin = gelu_f((in[g] - mean) * rstd * gm + bt);
    u = xin + tau * u * (1.f - o);
    o = (u - vth) > 0.f ? 1.f : 0.f;
    out[g] = o;
  }
}

// --------------------------- fused MLP branch ------------------------------
// xio(=x2) -> GN -> conv C->384 -> GELU -> conv 384->C -> += residual.
// 24 KB raw-staging region is reused as the 48 KB hidden buffer afterwards.
__global__ void mlp_fused_kernel(float* __restrict__ xio,
                                 const float* __restrict__ statsIn,
                                 const float* __restrict__ gamma, const float* __restrict__ beta,
                                 const float* __restrict__ w1, const float* __restrict__ b1,
                                 const float* __restrict__ w2, const float* __restrict__ b2)
{
  __shared__ __align__(16) unsigned char smem[61440];   // 60 KB
  ushort_t* sX    = (ushort_t*)smem;                    // 12 KB frag-ordered input
  float*    sRawF = (float*)(smem + 12288);             // 24 KB raw (reused below)
  ushort_t* sH    = (ushort_t*)(smem + 12288);          // 48 KB hidden frags

  const int tid = threadIdx.x;
  const int b  = blockIdx.y;
  const int p0 = blockIdx.x * 64;
  float mean, rstd; load_stats(statsIn, b, mean, rstd);
  __builtin_prefetch(w1, 0, 0);
  const size_t base = (size_t)b * NPB;

  stage_tile_async(xio + base + p0, (unsigned)(uintptr_t)sRawF, tid);
  wait_async0();
  __syncthreads();
  for (int idx = tid; idx < C_ * 64; idx += 128) {
    const int c = idx >> 6, p = idx & 63;
    const float v = (sRawF[idx] - mean) * rstd * gamma[c] + beta[c];
    const int kt = c >> 5, kloc = c & 31;
    const int lane = frag_half(kloc) * 16 + (p & 15);
    sX[((((p >> 4) * 3 + kt) * 32 + lane) << 4) + frag_e(kloc)] = f2bf(v);
  }
  __syncthreads();   // sRawF dead from here; region becomes sH

  const int wave = tid >> 5, lane = tid & 31;
  const int n = lane & 15, half = lane >> 4;
  ushort_t* sHw = sH + wave * 6144;      // per-wave [kt(12)][lane(32)][e(16)]

  // GEMM1: hidden = gelu(W1 @ xn + b1), 384 channels, written as B-fragments
  for (int mt = 0; mt < 24; ++mt) {
    v8f acc = {};
    for (int ks = 0; ks < 3; ++ks) {
      const v16bf af = load_a_frag_global(w1, mt * 16 + n, C_, ks * 32, half);
      const v16bf bf = load_frag_lds(sX, ((wave * 3 + ks) * 32 + lane) << 4);
      acc = __builtin_amdgcn_wmma_f32_16x16x32_bf16(false, af, false, bf,
                                                    (short)0, acc, false, false);
    }
    #pragma unroll
    for (int r = 0; r < 8; ++r) {
      const int hc = mt * 16 + half * 8 + r;
      const int kt2 = hc >> 5, kloc = hc & 31;
      const int lane2 = frag_half(kloc) * 16 + n;
      sHw[((kt2 * 32 + lane2) << 4) + frag_e(kloc)] = f2bf(gelu_f(acc[r] + b1[hc]));
    }
  }
  __syncthreads();

  // GEMM2: m = W2 @ hidden + b2 ; out = x2 + m   (12 K-steps of 32)
  for (int mt = 0; mt < 6; ++mt) {
    v8f acc = {};
    for (int ks = 0; ks < 12; ++ks) {
      const v16bf af = load_a_frag_global(w2, mt * 16 + n, HD_, ks * 32, half);
      const v16bf bf = load_frag_lds(sHw, (ks * 32 + lane) << 4);
      acc = __builtin_amdgcn_wmma_f32_16x16x32_bf16(false, af, false, bf,
                                                    (short)0, acc, false, false);
    }
    #pragma unroll
    for (int r = 0; r < 8; ++r) {
      const int oc = mt * 16 + half * 8 + r;
      const size_t g = base + (size_t)oc * HW_ + p0 + wave * 16 + n;
      xio[g] = xio[g] + acc[r] + b2[oc];
    }
  }
}

// ---------------------------------------------------------------------------
extern "C" void kernel_launch(void* const* d_in, const int* in_sizes, int n_in,
                              void* d_out, int out_size, void* d_ws, size_t ws_size,
                              hipStream_t stream)
{
  (void)in_sizes; (void)n_in; (void)out_size; (void)ws_size;
  const float* x    = (const float*)d_in[0];
  const float* n1g  = (const float*)d_in[1];
  const float* n1b  = (const float*)d_in[2];
  const float* c1w  = (const float*)d_in[3];
  const float* c1b  = (const float*)d_in[4];
  const float* g1g  = (const float*)d_in[5];
  const float* g1b  = (const float*)d_in[6];
  const float* dww  = (const float*)d_in[7];
  const float* dwb  = (const float*)d_in[8];
  const float* g2g  = (const float*)d_in[9];
  const float* g2b  = (const float*)d_in[10];
  const float* tau1 = (const float*)d_in[11];
  const float* vth1 = (const float*)d_in[12];
  const float* tau2 = (const float*)d_in[13];
  const float* vth2 = (const float*)d_in[14];
  const float* c21w = (const float*)d_in[15];
  const float* c21b = (const float*)d_in[16];
  const float* c22w = (const float*)d_in[17];
  const float* c22b = (const float*)d_in[18];
  const float* g3g  = (const float*)d_in[19];
  const float* g3b  = (const float*)d_in[20];
  const float* c3w  = (const float*)d_in[21];
  const float* c3b  = (const float*)d_in[22];
  const float* n2g  = (const float*)d_in[23];
  const float* n2b  = (const float*)d_in[24];
  const float* f1w  = (const float*)d_in[25];
  const float* f1b  = (const float*)d_in[26];
  const float* f2w  = (const float*)d_in[27];
  const float* f2b  = (const float*)d_in[28];
  float* out = (float*)d_out;

  float* stats = (float*)d_ws;      // 5 stat buffers, each B_*2 floats
  float* s0 = stats + 0 * 128;
  float* s1 = stats + 1 * 128;
  float* s2 = stats + 2 * 128;
  float* s3 = stats + 3 * 128;
  float* s4 = stats + 4 * 128;
  const size_t NB = (size_t)B_ * NPB;
  float* bufA = stats + 1024;
  float* bufB = bufA + NB;
  float* bufC = bufB + NB;

  const dim3 blk128(128), blk256(256);
  const dim3 gConv(49, B_), gElem(1176, B_), gLif(1344);

  // 0) zero stats accumulators (graph-replay safe: runs every launch)
  hipLaunchKernelGGL(zero_stats_kernel, dim3(1), blk256, 0, stream, stats, 640);
  // 1) per-batch stats of x (for norm1)
  hipLaunchKernelGGL(batch_stats_kernel, gElem, blk256, 0, stream, x, s0);
  // 2) h1 = conv1(GN(x)) ; stats(h1) -> s1
  hipLaunchKernelGGL((conv1x1_wmma_kernel<true, false, false, true>), gConv, blk128, 0, stream,
                     x, c1w, c1b, nullptr, nullptr, nullptr, s0, n1g, n1b, nullptr, bufA, s1);
  // 3) h3 = dwconv3(gelu(GN(h1))) ; stats(h3) -> s2
  hipLaunchKernelGGL(dwconv3_kernel, gElem, blk256, 0, stream,
                     bufA, dww, dwb, s1, g1g, g1b, bufB, s2);
  // 4) x_lr = LIF over H of gelu(GN(h3)) ; x_td = LIF over W
  hipLaunchKernelGGL((lif_kernel<true>),  gLif, blk256, 0, stream,
                     bufB, s2, g2g, g2b, tau1, vth1, bufA);
  hipLaunchKernelGGL((lif_kernel<false>), gLif, blk256, 0, stream,
                     bufB, s2, g2g, g2b, tau2, vth2, bufC);
  // 5) h5 = gelu(conv(x_lr,c21)) + gelu(conv(x_td,c22)) ; stats -> s3
  hipLaunchKernelGGL((conv1x1_wmma_kernel<false, true, false, true>), gConv, blk128, 0, stream,
                     bufA, c21w, c21b, bufC, c22w, c22b, nullptr, nullptr, nullptr, nullptr, bufB, s3);
  // 6) x2 = x + conv(GN(h5), c3) -> d_out ; stats(x2) -> s4
  hipLaunchKernelGGL((conv1x1_wmma_kernel<true, false, true, true>), gConv, blk128, 0, stream,
                     bufB, c3w, c3b, nullptr, nullptr, nullptr, s3, g3g, g3b, x, out, s4);
  // 7) out = x2 + MLP(GN(x2))   (fused, hidden kept in LDS)
  hipLaunchKernelGGL(mlp_fused_kernel, gConv, blk128, 0, stream,
                     out, s4, n2g, n2b, f1w, f1b, f2w, f2b);
}
